// EGNN_Layer_17282948399520
// MI455X (gfx1250) — compile-verified, matrix operand order checked by reading, not run on hardware
//
#include <hip/hip_runtime.h>

typedef __attribute__((ext_vector_type(16))) _Float16 v16h;
typedef __attribute__((ext_vector_type(8)))  float    v8f;

// 16-bit A-matrix (16x32) WMMA layout: lane L holds row m=L%16; half index j
// (VGPR v=j/2, hi/lo h=j&1) maps to K = 16*(v/4) + 8*(L/16) + 2*(v&3) + h.
__device__ __forceinline__ int kmap(int g, int j) {
    int v = j >> 1, h = j & 1;
    return ((v >> 2) << 4) | (g << 3) | ((v & 3) << 1) | h;
}

__device__ __forceinline__ float silu_f(float v) {
    return v / (1.0f + __expf(-v));
}

__device__ __forceinline__ v16h load_afrag(const _Float16* p, int g) {
    v16h a;
#pragma unroll
    for (int j = 0; j < 16; ++j) a[j] = p[kmap(g, j)];
    return a;
}

// B fragments prepacked in ws: chunk (nt,kt,lane) -> 16 contiguous halves (32B)
__device__ __forceinline__ v16h load_bfrag(const _Float16* base, int nt, int kt,
                                           int KT, int lane) {
    return *(const v16h*)(base + ((size_t)((nt * KT + kt) * 32 + lane) << 4));
}

// ---------------------------------------------------------------------------
// Weight repack: W is [Kreal x 64] f32 row-major; emit f16 B-fragments for
// B^T-as-A-layout. One thread per (nt, kt, lane), writes 16 halves.
__global__ void prep_frag(const float* __restrict__ W, _Float16* __restrict__ dst,
                          int Kreal, int KT) {
    int t = blockIdx.x * blockDim.x + threadIdx.x;
    int total = 4 * KT * 32;
    if (t >= total) return;
    int lane = t & 31;
    int kt = (t >> 5) % KT;
    int nt = (t >> 5) / KT;
    int g = lane >> 4;
    int n = nt * 16 + (lane & 15);
    _Float16* out = dst + ((size_t)((nt * KT + kt) * 32 + lane) << 4);
#pragma unroll
    for (int j = 0; j < 16; ++j) {
        int k = kt * 32 + kmap(g, j);
        out[j] = (k < Kreal) ? (_Float16)W[(size_t)k * 64 + n] : (_Float16)0.0f;
    }
}

__global__ void zero_kernel(float* __restrict__ p, long n) {
    long i = (long)blockIdx.x * blockDim.x + threadIdx.x;
    long stride = (long)gridDim.x * blockDim.x;
    for (; i < n; i += stride) p[i] = 0.0f;
}

// ---------------------------------------------------------------------------
// Edge kernel: one wave handles 16 edges. 8 waves/block, private LDS regions.
__global__ void __launch_bounds__(256) egnn_edge_kernel(
    const float* __restrict__ x, const float* __restrict__ h,
    const float* __restrict__ ef,
    const float* __restrict__ b1e, const float* __restrict__ b2e,
    const float* __restrict__ bc1, const float* __restrict__ wc2,
    const float* __restrict__ bc2,
    const int* __restrict__ row, const int* __restrict__ col,
    const _Float16* __restrict__ W1f, const _Float16* __restrict__ W2f,
    const _Float16* __restrict__ Wc1f,
    float* __restrict__ tot_f, float* __restrict__ deg,
    float* __restrict__ tot_msg, int M)
{
    __shared__ __align__(16) unsigned char sRaw[8][5120]; // 16x160 halves / 16x64 floats
    __shared__ int   sRowA[8][16];
    __shared__ int   sColA[8][16];
    __shared__ float sSq[8][16];

    const int tid  = threadIdx.x;
    const int wave = tid >> 5;
    const int lane = tid & 31;
    const int g    = lane >> 4;
    const int m    = lane & 15;
    const long e0  = ((long)blockIdx.x * 8 + wave) * 16;

    const long e  = e0 + m;
    const bool ev = e < (long)M;
    const long ec = ev ? e : (long)(M - 1);
    const int r = row[ec], c = col[ec];
    const float rx = x[3 * (long)r + 0] - x[3 * (long)c + 0];
    const float ry = x[3 * (long)r + 1] - x[3 * (long)c + 1];
    const float rz = x[3 * (long)r + 2] - x[3 * (long)c + 2];
    const float sq = rx * rx + ry * ry + rz * rz;
    if (g == 0) { sRowA[wave][m] = r; sColA[wave][m] = c; sSq[wave][m] = sq; }

    _Float16* S = (_Float16*)sRaw[wave];
    // gather s = [|rij|^2, h[row], h[col], edge_fea, 0-pad] -> 16x160 f16 tile
    for (int i = 0; i < 80; ++i) {
        int t = lane + 32 * i;
        int mm = t / 160;
        int cc = t - mm * 160;
        long et = e0 + mm;
        float val = 0.0f;
        if (et < (long)M) {
            if (cc == 0)       val = sSq[wave][mm];
            else if (cc < 65)  val = h[(long)sRowA[wave][mm] * 64 + (cc - 1)];
            else if (cc < 129) val = h[(long)sColA[wave][mm] * 64 + (cc - 65)];
            else if (cc < 145) val = ef[et * 16 + (cc - 129)];
        }
        S[t] = (_Float16)val;
    }

    // ---- layer 1: [16x160] x [160x64] ----
    v16h a1[5];
#pragma unroll
    for (int kt = 0; kt < 5; ++kt) a1[kt] = load_afrag(S + m * 160 + kt * 32, g);

    v8f acc[4];
#pragma unroll
    for (int nt = 0; nt < 4; ++nt) {
        float b = b1e[nt * 16 + m];
#pragma unroll
        for (int i = 0; i < 8; ++i) acc[nt][i] = b;
    }
#pragma unroll
    for (int kt = 0; kt < 5; ++kt)
#pragma unroll
        for (int nt = 0; nt < 4; ++nt) {
            v16h bf = load_bfrag(W1f, nt, kt, 5, lane);
            acc[nt] = __builtin_amdgcn_wmma_f32_16x16x32_f16(
                false, a1[kt], false, bf, (short)0, acc[nt], false, false);
        }
    // silu + restage (C layout -> LDS 16x64 halves -> A layout)
#pragma unroll
    for (int nt = 0; nt < 4; ++nt)
#pragma unroll
        for (int v = 0; v < 8; ++v)
            S[(v + 8 * g) * 64 + nt * 16 + m] = (_Float16)silu_f(acc[nt][v]);

    // ---- layer 2: [16x64] x [64x64] ----
    v16h a2[2];
    a2[0] = load_afrag(S + m * 64 + 0, g);
    a2[1] = load_afrag(S + m * 64 + 32, g);

    v8f msg[4];
#pragma unroll
    for (int nt = 0; nt < 4; ++nt) {
        float b = b2e[nt * 16 + m];
#pragma unroll
        for (int i = 0; i < 8; ++i) msg[nt][i] = b;
    }
#pragma unroll
    for (int kt = 0; kt < 2; ++kt)
#pragma unroll
        for (int nt = 0; nt < 4; ++nt) {
            v16h bf = load_bfrag(W2f, nt, kt, 2, lane);
            msg[nt] = __builtin_amdgcn_wmma_f32_16x16x32_f16(
                false, a2[kt], false, bf, (short)0, msg[nt], false, false);
        }
#pragma unroll
    for (int nt = 0; nt < 4; ++nt)
#pragma unroll
        for (int v = 0; v < 8; ++v) msg[nt][v] = silu_f(msg[nt][v]);

    // ---- coord layer 1: [16x64] x [64x64] ----
#pragma unroll
    for (int nt = 0; nt < 4; ++nt)
#pragma unroll
        for (int v = 0; v < 8; ++v)
            S[(v + 8 * g) * 64 + nt * 16 + m] = (_Float16)msg[nt][v];

    v16h a3[2];
    a3[0] = load_afrag(S + m * 64 + 0, g);
    a3[1] = load_afrag(S + m * 64 + 32, g);

    v8f c1[4];
#pragma unroll
    for (int nt = 0; nt < 4; ++nt) {
        float b = bc1[nt * 16 + m];
#pragma unroll
        for (int i = 0; i < 8; ++i) c1[nt][i] = b;
    }
#pragma unroll
    for (int kt = 0; kt < 2; ++kt)
#pragma unroll
        for (int nt = 0; nt < 4; ++nt) {
            v16h bf = load_bfrag(Wc1f, nt, kt, 2, lane);
            c1[nt] = __builtin_amdgcn_wmma_f32_16x16x32_f16(
                false, a3[kt], false, bf, (short)0, c1[nt], false, false);
        }

    // ---- coord layer 2: [16x64] x [64x1] via LDS row reduce ----
    float* SF = (float*)sRaw[wave];
#pragma unroll
    for (int nt = 0; nt < 4; ++nt) {
        float wv = wc2[nt * 16 + m];
#pragma unroll
        for (int v = 0; v < 8; ++v)
            SF[(v + 8 * g) * 64 + nt * 16 + m] = silu_f(c1[nt][v]) * wv;
    }
    float part = 0.0f;
#pragma unroll 8
    for (int n = 0; n < 32; ++n) part += SF[m * 64 + g * 32 + n];
    part += __shfl_xor(part, 16, 32);
    const float cm = part + bc2[0];

    if (g == 0 && ev) {
        atomicAdd(&tot_f[3 * (long)r + 0], rx * cm);
        atomicAdd(&tot_f[3 * (long)r + 1], ry * cm);
        atomicAdd(&tot_f[3 * (long)r + 2], rz * cm);
        atomicAdd(&deg[r], 1.0f);
    }
    // segment-sum of message per row node
#pragma unroll
    for (int nt = 0; nt < 4; ++nt)
#pragma unroll
        for (int v = 0; v < 8; ++v) {
            int mr = v + 8 * g;
            if (e0 + mr < (long)M) {
                int rr = sRowA[wave][mr];
                atomicAdd(&tot_msg[(long)rr * 64 + nt * 16 + m], msg[nt][v]);
            }
        }
}

// ---------------------------------------------------------------------------
// Node kernel: one wave handles 16 nodes. Reads tot_msg from hout, overwrites
// hout with h_new; updates xout in place from tot_f/deg.
__global__ void __launch_bounds__(256) egnn_node_kernel(
    const float* __restrict__ x, const float* __restrict__ h,
    const float* __restrict__ bn1, const float* __restrict__ bn2,
    const _Float16* __restrict__ Wn1f, const _Float16* __restrict__ Wn2f,
    const float* __restrict__ deg,
    float* __restrict__ xout, float* __restrict__ hout, int N)
{
    __shared__ __align__(16) unsigned char sRaw[8][4096]; // 16x128 halves

    const int tid  = threadIdx.x;
    const int wave = tid >> 5;
    const int lane = tid & 31;
    const int g    = lane >> 4;
    const int m    = lane & 15;
    const long n0  = ((long)blockIdx.x * 8 + wave) * 16;
    const long nid = n0 + m;

    if (g == 0 && nid < (long)N) {
        float dd = deg[nid];
        dd = dd > 1.0f ? dd : 1.0f;
#pragma unroll
        for (int d = 0; d < 3; ++d) {
            float tf = xout[nid * 3 + d] / dd;
            tf = tf > 100.0f ? 100.0f : (tf < -100.0f ? -100.0f : tf);
            xout[nid * 3 + d] = x[nid * 3 + d] + tf;
        }
    }

    _Float16* S = (_Float16*)sRaw[wave];
    for (int i = 0; i < 64; ++i) {
        int t = lane + 32 * i;
        int mm = t >> 7, cc = t & 127;
        long nn = n0 + mm;
        float val = 0.0f;
        if (nn < (long)N)
            val = (cc < 64) ? h[nn * 64 + cc] : hout[nn * 64 + (cc - 64)];
        S[t] = (_Float16)val;
    }

    v16h a[4];
#pragma unroll
    for (int kt = 0; kt < 4; ++kt) a[kt] = load_afrag(S + m * 128 + kt * 32, g);

    v8f acc[4];
#pragma unroll
    for (int nt = 0; nt < 4; ++nt) {
        float b = bn1[nt * 16 + m];
#pragma unroll
        for (int i = 0; i < 8; ++i) acc[nt][i] = b;
    }
#pragma unroll
    for (int kt = 0; kt < 4; ++kt)
#pragma unroll
        for (int nt = 0; nt < 4; ++nt) {
            v16h bf = load_bfrag(Wn1f, nt, kt, 4, lane);
            acc[nt] = __builtin_amdgcn_wmma_f32_16x16x32_f16(
                false, a[kt], false, bf, (short)0, acc[nt], false, false);
        }
#pragma unroll
    for (int nt = 0; nt < 4; ++nt)
#pragma unroll
        for (int v = 0; v < 8; ++v)
            S[(v + 8 * g) * 64 + nt * 16 + m] = (_Float16)silu_f(acc[nt][v]);

    v16h a2[2];
    a2[0] = load_afrag(S + m * 64 + 0, g);
    a2[1] = load_afrag(S + m * 64 + 32, g);

    v8f o[4];
#pragma unroll
    for (int nt = 0; nt < 4; ++nt) {
        float b = bn2[nt * 16 + m];
#pragma unroll
        for (int i = 0; i < 8; ++i) o[nt][i] = b;
    }
#pragma unroll
    for (int kt = 0; kt < 2; ++kt)
#pragma unroll
        for (int nt = 0; nt < 4; ++nt) {
            v16h bf = load_bfrag(Wn2f, nt, kt, 2, lane);
            o[nt] = __builtin_amdgcn_wmma_f32_16x16x32_f16(
                false, a2[kt], false, bf, (short)0, o[nt], false, false);
        }
#pragma unroll
    for (int nt = 0; nt < 4; ++nt)
#pragma unroll
        for (int v = 0; v < 8; ++v) {
            long nn = n0 + v + 8 * g;
            if (nn < (long)N) hout[nn * 64 + nt * 16 + m] = o[nt][v];
        }
}

// ---------------------------------------------------------------------------
extern "C" void kernel_launch(void* const* d_in, const int* in_sizes, int n_in,
                              void* d_out, int out_size, void* d_ws, size_t ws_size,
                              hipStream_t stream) {
    (void)n_in; (void)out_size; (void)ws_size;
    const float* x   = (const float*)d_in[0];
    const float* h   = (const float*)d_in[1];
    const float* ef  = (const float*)d_in[2];
    const float* w1e = (const float*)d_in[3];
    const float* b1e = (const float*)d_in[4];
    const float* w2e = (const float*)d_in[5];
    const float* b2e = (const float*)d_in[6];
    const float* wc1 = (const float*)d_in[7];
    const float* bc1 = (const float*)d_in[8];
    const float* wc2 = (const float*)d_in[9];
    const float* bc2 = (const float*)d_in[10];
    const float* wn1 = (const float*)d_in[11];
    const float* bn1 = (const float*)d_in[12];
    const float* wn2 = (const float*)d_in[13];
    const float* bn2 = (const float*)d_in[14];
    const int*   row = (const int*)d_in[15];
    const int*   col = (const int*)d_in[16];

    const int N = in_sizes[0] / 3;
    const int M = in_sizes[15];

    float* xout = (float*)d_out;              // tot_f accumulator then x_new
    float* hout = xout + (size_t)3 * N;       // tot_msg accumulator then h_new

    // workspace: weight fragments (halves) then deg (floats)
    _Float16* W1f  = (_Float16*)d_ws;         // 160x64 -> 10240 halves
    _Float16* W2f  = W1f + 10240;             //  64x64 ->  4096
    _Float16* Wc1f = W2f + 4096;              //  64x64 ->  4096
    _Float16* Wn1f = Wc1f + 4096;             // 128x64 ->  8192
    _Float16* Wn2f = Wn1f + 8192;             //  64x64 ->  4096
    float* deg = (float*)((char*)d_ws + 65536);

    // zero accumulators (deterministic each launch; graph-capture safe)
    zero_kernel<<<1024, 256, 0, stream>>>(xout, (long)N * 67);
    zero_kernel<<<256, 256, 0, stream>>>(deg, (long)N);

    prep_frag<<<5, 128, 0, stream>>>(w1e, W1f, 145, 5);
    prep_frag<<<2, 128, 0, stream>>>(w2e, W2f, 64, 2);
    prep_frag<<<2, 128, 0, stream>>>(wc1, Wc1f, 64, 2);
    prep_frag<<<4, 128, 0, stream>>>(wn1, Wn1f, 128, 4);
    prep_frag<<<2, 128, 0, stream>>>(wn2, Wn2f, 64, 2);

    const int etiles = (M + 15) / 16;
    const int eblocks = (etiles + 7) / 8;
    egnn_edge_kernel<<<eblocks, 256, 0, stream>>>(
        x, h, ef, b1e, b2e, bc1, wc2, bc2, row, col,
        W1f, W2f, Wc1f, xout, deg, hout, M);

    const int ntiles = (N + 15) / 16;
    const int nblocks = (ntiles + 7) / 8;
    egnn_node_kernel<<<nblocks, 256, 0, stream>>>(
        x, h, bn1, bn2, Wn1f, Wn2f, deg, xout, hout, N);
}
